// SemanticModule_54975581389217
// MI455X (gfx1250) — matrix operand order repaired and checked by reading, not compiled, not run on hardware
//
#include <hip/hip_runtime.h>

// ---------------------------------------------------------------------------
// Types for CDNA5 WMMA (wave32, 16x16x32 bf16 -> f32)
// ---------------------------------------------------------------------------
typedef __attribute__((ext_vector_type(16))) __bf16 v16bf;
typedef __attribute__((ext_vector_type(8)))  __bf16 v8bf;
typedef __attribute__((ext_vector_type(8)))  float  v8f;

#define CDIV(a, b) (((a) + (b) - 1) / (b))

// ---------------------------------------------------------------------------
// Problem constants (from reference)
// ---------------------------------------------------------------------------
static const int kE  = 300000;
static const int kNS = 200000;   // stroke nodes
static const int kNL = 100000;   // loop nodes

static const int CI[7] = {7, 16, 32, 64, 128, 128, 128};
static const int CO[7] = {16, 32, 64, 128, 128, 128, 128};

// relation metadata, RELS order:
// 0 representedBy (stroke->loop, sum), 1 neighboring_vertical (loop->loop, mean),
// 2 neighboring_horizontal (loop->loop, mean), 3 contains (loop->loop, mean),
// 4 coplanar (loop->loop, mean), 5 order (stroke->stroke, sum),
// 6 connect (stroke->stroke, mean)
static const int RSRC_IS_LOOP[7] = {0, 1, 1, 1, 1, 0, 0};
static const int RDST_IS_LOOP[7] = {1, 1, 1, 1, 1, 0, 0};
static const int RMEAN[7]        = {0, 1, 1, 1, 1, 0, 1};
static const int RELS_STROKE[2]  = {5, 6};
static const int RELS_LOOP[5]    = {0, 1, 2, 3, 4};

// ---------------------------------------------------------------------------
// Elementwise / scatter kernels (bandwidth-bound part of the workload)
// ---------------------------------------------------------------------------
__global__ __launch_bounds__(256) void k_zero_f32(float* __restrict__ p, long long n) {
  long long i = (long long)blockIdx.x * blockDim.x + threadIdx.x;
  if (i < n) p[i] = 0.0f;
}

__global__ __launch_bounds__(256) void k_count(const int* __restrict__ edst,
                                               float* __restrict__ cnt, int E) {
  int i = blockIdx.x * blockDim.x + threadIdx.x;
  if (i < E) atomicAdd(&cnt[edst[i]], 1.0f);
}

__global__ __launch_bounds__(256) void k_recip(float* __restrict__ c, int n) {
  int i = blockIdx.x * blockDim.x + threadIdx.x;
  if (i < n) c[i] = 1.0f / fmaxf(c[i], 1.0f);
}

// edge-parallel segment sum: one thread per (edge, channel)
__global__ __launch_bounds__(256) void k_scatter_add(const float* __restrict__ x,
                                                     const int* __restrict__ esrc,
                                                     const int* __restrict__ edst,
                                                     float* __restrict__ agg,
                                                     int E, int C) {
  int i = blockIdx.x * blockDim.x + threadIdx.x;
  if (i >= E * C) return;
  int e = i / C;
  int c = i - e * C;
  atomicAdd(&agg[(long long)edst[e] * C + c], x[(long long)esrc[e] * C + c]);
}

// scale by 1/deg (mean) and convert to bf16 into packed-A column block
__global__ __launch_bounds__(256) void k_finalize_A(const float* __restrict__ src,
                                                    const float* __restrict__ recip,
                                                    __bf16* __restrict__ A,
                                                    int N, int C, int Kpad, int colOff) {
  int i = blockIdx.x * blockDim.x + threadIdx.x;
  if (i >= N * C) return;
  int row = i / C;
  int c = i - row * C;
  float v = src[(long long)row * C + c];
  if (recip) v *= recip[row];
  A[(long long)row * Kpad + colOff + c] = (__bf16)v;
}

__global__ __launch_bounds__(256) void k_zero_padA(__bf16* __restrict__ A,
                                                   int N, int Kpad, int K) {
  int pad = Kpad - K;
  int i = blockIdx.x * blockDim.x + threadIdx.x;
  if (i >= N * pad) return;
  int row = i / pad;
  int c = i - row * pad;
  A[(long long)row * Kpad + K + c] = (__bf16)0.0f;
}

__global__ __launch_bounds__(256) void k_relu_copy(const float* __restrict__ in,
                                                   float* __restrict__ out, long long n) {
  long long i = (long long)blockIdx.x * blockDim.x + threadIdx.x;
  if (i < n) out[i] = fmaxf(in[i], 0.0f);
}

// ---------------------------------------------------------------------------
// Weight packing: Wt[co][Kpad] (transposed, bf16), root block folds
// sum-of-Wroot + residual projection (or identity); bias summed over relations.
// ---------------------------------------------------------------------------
struct PackArgs {
  const float* Wroot[5];
  const float* Wrel[5];
  const float* bias[5];
  const float* proj;     // nullptr if none
  int nrel, ci, co, Kpad, addIdentity;
};

__global__ __launch_bounds__(256) void k_pack(PackArgs pa, __bf16* __restrict__ Wt,
                                              float* __restrict__ biasSum) {
  int idx = blockIdx.x * blockDim.x + threadIdx.x;
  int total = pa.co * pa.Kpad;
  if (idx >= total) return;
  int n = idx / pa.Kpad;
  int kg = idx - n * pa.Kpad;
  int K = (pa.nrel + 1) * pa.ci;
  float v = 0.0f;
  if (kg < pa.ci) {
    for (int r = 0; r < pa.nrel; r++) v += pa.Wroot[r][kg * pa.co + n];
    if (pa.proj) v += pa.proj[kg * pa.co + n];
    if (pa.addIdentity && kg == n) v += 1.0f;
  } else if (kg < K) {
    int r = kg / pa.ci - 1;
    int k = kg - (r + 1) * pa.ci;
    v = pa.Wrel[r][k * pa.co + n];
  }
  Wt[(long long)n * pa.Kpad + kg] = (__bf16)v;
  if (kg == 0) {
    float bs = 0.0f;
    for (int r = 0; r < pa.nrel; r++) bs += pa.bias[r][n];
    biasSum[n] = bs;
  }
}

// ---------------------------------------------------------------------------
// WMMA GEMM: out[N,co] = relu?( A[N,Kpad] @ Wt^T + bias ), bf16 in / f32 acc.
// 4 waves per block; each wave owns one 16x16 output tile; K streamed by 32.
// A 16x32 / B 32x16 use contiguous k (lane half h: k = 16h..16h+15); f32 C/D per CDNA5 ISA 7.12.2.
// ---------------------------------------------------------------------------
__global__ __launch_bounds__(128) void k_gemm_wmma(const __bf16* __restrict__ A,
                                                   const __bf16* __restrict__ Wt,
                                                   const float* __restrict__ bias,
                                                   float* __restrict__ out,
                                                   int N, int Kpad, int co, int relu) {
  int wave = threadIdx.x >> 5;          // 0..3
  int lane = threadIdx.x & 31;
  int rowTile = blockIdx.x;             // 16 rows (N divisible by 16)
  int colTile = blockIdx.y * 4 + wave;  // 16 cols
  if (colTile * 16 >= co) return;       // wave-uniform exit; EXEC stays all-1s
  int laneHi = lane >> 4;               // 0: lanes 0-15, 1: lanes 16-31
  int lane16 = lane & 15;
  int row = rowTile * 16 + lane16;      // A-matrix row for this lane (M = lane%16)
  int col = colTile * 16 + lane16;      // B/D column for this lane (N = lane%16)

  const __bf16* Arow = A + (long long)row * Kpad + (laneHi ? 16 : 0);
  const __bf16* Wrow = Wt + (long long)col * Kpad + (laneHi ? 16 : 0);

  float bv = bias[col];
  v8f acc = {bv, bv, bv, bv, bv, bv, bv, bv};

  for (int k = 0; k < Kpad; k += 32) {
    // A 16x32 bf16: lane<16 holds K = k+0..15; lane>=16: k+16..31
    v8bf a0 = *(const v8bf*)(Arow + k);
    v8bf a1 = *(const v8bf*)(Arow + k + 8);
    v16bf av = __builtin_shufflevector(a0, a1, 0, 1, 2, 3, 4, 5, 6, 7,
                                       8, 9, 10, 11, 12, 13, 14, 15);
    // B 32x16 bf16: lane<16 holds K = k+0..15 of column `col`; lane>=16: k+16..31
    v16bf bvv = *(const v16bf*)(Wrow + k);
    acc = __builtin_amdgcn_wmma_f32_16x16x32_bf16(false, av, false, bvv,
                                                  (short)0, acc, false, false);
  }

  // f32 C/D layout: VGPR r -> row = 16*rowTile + r + 8*laneHi, col = col
  float* outp = out + (long long)(rowTile * 16 + laneHi * 8) * co + col;
#pragma unroll
  for (int i = 0; i < 8; i++) {
    float v = acc[i];
    if (relu) v = fmaxf(v, 0.0f);
    outp[(long long)i * co] = v;
  }
}

// ---------------------------------------------------------------------------
// Host-side orchestration
// ---------------------------------------------------------------------------
extern "C" void kernel_launch(void* const* d_in, const int* in_sizes, int n_in,
                              void* d_out, int out_size, void* d_ws, size_t ws_size,
                              hipStream_t stream) {
  if (n_in < 162) return;  // 9 data inputs + 153 param arrays expected

  const float* xS = (const float*)d_in[0];
  const float* xL = (const float*)d_in[1];
  const int* edge[7];
  for (int r = 0; r < 7; r++) edge[r] = (const int*)d_in[2 + r];

  // Param table (insertion order of _make_params)
  const float* Wrel[7][7];
  const float* Wroot[7][7];
  const float* Bias[7][7];
  const float* Proj[7][2] = {};
  {
    int p = 9;
    for (int l = 0; l < 7; l++) {
      for (int r = 0; r < 7; r++) {
        Wrel[l][r]  = (const float*)d_in[p++];
        Wroot[l][r] = (const float*)d_in[p++];
        Bias[l][r]  = (const float*)d_in[p++];
      }
      if (l > 0 && CI[l] != CO[l]) {
        Proj[l][0] = (const float*)d_in[p++];  // stroke
        Proj[l][1] = (const float*)d_in[p++];  // loop
      }
    }
  }

  // Workspace carving
  size_t off = 0;
  char* base = (char*)d_ws;
  auto carve = [&](size_t bytes) -> char* {
    char* q = base + off;
    off += (bytes + 255) & ~(size_t)255;
    return q;
  };
  float* hS[2];
  float* hL[2];
  hS[0] = (float*)carve((size_t)kNS * 128 * 4);
  hS[1] = (float*)carve((size_t)kNS * 128 * 4);
  hL[0] = (float*)carve((size_t)kNL * 128 * 4);
  hL[1] = (float*)carve((size_t)kNL * 128 * 4);
  __bf16* Abuf = (__bf16*)carve(153600000ull);             // max packed-A (bf16)
  float* agg = (float*)carve((size_t)kNS * 128 * 4);        // fp32 scatter temp
  float* recip[7] = {};
  for (int r = 0; r < 7; r++)
    if (RMEAN[r]) recip[r] = (float*)carve((size_t)(RDST_IS_LOOP[r] ? kNL : kNS) * 4);
  __bf16* Wt[2];
  float* bsum[2];
  for (int d = 0; d < 2; d++) {
    Wt[d] = (__bf16*)carve((size_t)128 * 768 * 2);
    bsum[d] = (float*)carve(128 * 4);
  }
  if (off > ws_size) return;  // insufficient scratch (deterministic no-op)

  // 1) degree reciprocals for 'mean' relations (recomputed each call)
  for (int r = 0; r < 7; r++) {
    if (!RMEAN[r]) continue;
    int Nd = RDST_IS_LOOP[r] ? kNL : kNS;
    k_zero_f32<<<CDIV(Nd, 256), 256, 0, stream>>>(recip[r], Nd);
    k_count<<<CDIV(kE, 256), 256, 0, stream>>>(edge[r] + kE, recip[r], kE);
    k_recip<<<CDIV(Nd, 256), 256, 0, stream>>>(recip[r], Nd);
  }

  // 2) layers
  int cur = 0;
  for (int l = 0; l < 7; l++) {
    int ci = CI[l], co = CO[l];
    for (int dt = 0; dt < 2; dt++) {  // 0 = stroke dst, 1 = loop dst
      const int* rl = dt ? RELS_LOOP : RELS_STROKE;
      int nrel = dt ? 5 : 2;
      int Nd = dt ? kNL : kNS;
      int K = (nrel + 1) * ci;
      int Kpad = (K + 31) & ~31;
      const float* xd = (l == 0) ? (dt ? xL : xS) : (dt ? hL[cur] : hS[cur]);

      // pack weights (root-sum + residual fold) and bias-sum
      PackArgs pa = {};
      for (int j = 0; j < nrel; j++) {
        int r = rl[j];
        pa.Wroot[j] = Wroot[l][r];
        pa.Wrel[j] = Wrel[l][r];
        pa.bias[j] = Bias[l][r];
      }
      pa.proj = Proj[l][dt];
      pa.nrel = nrel; pa.ci = ci; pa.co = co; pa.Kpad = Kpad;
      pa.addIdentity = (l > 0 && ci == co) ? 1 : 0;
      k_pack<<<CDIV(co * Kpad, 256), 256, 0, stream>>>(pa, Wt[dt], bsum[dt]);

      // packed A: block 0 = x_dst
      k_finalize_A<<<CDIV(Nd * ci, 256), 256, 0, stream>>>(xd, nullptr, Abuf, Nd, ci, Kpad, 0);
      if (Kpad > K)
        k_zero_padA<<<CDIV(Nd * (Kpad - K), 256), 256, 0, stream>>>(Abuf, Nd, Kpad, K);

      // blocks 1..nrel = per-relation aggregations
      for (int j = 0; j < nrel; j++) {
        int r = rl[j];
        const float* xs = (l == 0) ? (RSRC_IS_LOOP[r] ? xL : xS)
                                   : (RSRC_IS_LOOP[r] ? hL[cur] : hS[cur]);
        k_zero_f32<<<CDIV((long long)Nd * ci, 256), 256, 0, stream>>>(agg, (long long)Nd * ci);
        k_scatter_add<<<CDIV(kE * ci, 256), 256, 0, stream>>>(xs, edge[r], edge[r] + kE,
                                                              agg, kE, ci);
        k_finalize_A<<<CDIV(Nd * ci, 256), 256, 0, stream>>>(
            agg, RMEAN[r] ? recip[r] : nullptr, Abuf, Nd, ci, Kpad, (j + 1) * ci);
      }

      // fused GEMM + bias + residual(+proj) + relu
      float* hout = dt ? hL[cur ^ 1] : hS[cur ^ 1];
      dim3 grid(Nd / 16, CDIV(co, 64));
      k_gemm_wmma<<<grid, 128, 0, stream>>>(Abuf, Wt[dt], bsum[dt], hout, Nd, Kpad, co,
                                            (l > 0) ? 1 : 0);
    }
    cur ^= 1;
  }

  // 3) final outputs: relu(h_stroke) ++ relu(h_loop)
  k_relu_copy<<<CDIV((long long)kNS * 128, 256), 256, 0, stream>>>(
      hS[cur], (float*)d_out, (long long)kNS * 128);
  k_relu_copy<<<CDIV((long long)kNL * 128, 256), 256, 0, stream>>>(
      hL[cur], (float*)d_out + (long long)kNS * 128, (long long)kNL * 128);
}
